// KernelizedHeadAttention_47510928228692
// MI455X (gfx1250) — matrix operand order repaired and checked by reading, not compiled, not run on hardware
//
#include <hip/hip_runtime.h>
#include <hip/hip_bf16.h>

// ---------------------------------------------------------------------------
// KernelizedHeadAttention for MI455X (gfx1250, wave32, WMMA)
// B=1, S=2048, D=2048, H=16, DH=128, DHID=256, DKER=128
// ---------------------------------------------------------------------------

typedef __attribute__((ext_vector_type(16))) _Float16 v16h;
typedef __attribute__((ext_vector_type(8)))  float    v8f;

#define S_LEN 2048
#define NHEAD 16
#define DHEAD 128
#define DHID  256

union Frag {
  v16h  v;
  uint4 q[2];
};

__device__ __forceinline__ v8f wmma32(v16h a, v16h b, v8f c) {
  // D = A(16x32 f16) * B(32x16 f16) + C(16x16 f32)
  return __builtin_amdgcn_wmma_f32_16x16x32_f16(
      /*neg_a=*/false, a, /*neg_b=*/false, b,
      /*c_mod=*/(short)0, c, /*reuse_a=*/false, /*reuse_b=*/false);
}

// A fragment: row-major f16 row `row` (one row per lane, M = lane&15).
// ISA 16-bit A 16x32 layout: lanes g=lane>>4; halfs e<8 -> K=kb+8g+e,
// halfs e>=8 -> K=kb+16+8g+(e-8). Two 16B loads.
__device__ __forceinline__ v16h load_a(const _Float16* row, int kb, int grp) {
  Frag f;
  f.q[0] = *(const uint4*)(row + kb + 8 * grp);
  f.q[1] = *(const uint4*)(row + kb + 16 + 8 * grp);
  return f.v;
}

// B fragment from B^T stored [N][K] row-major f16: lane holds column
// N = lane&15; halfs e<8 -> K = kb + 8*(lane>>4) + e, e>=8 -> K = kb + 16 + 8*(lane>>4) + (e-8).
__device__ __forceinline__ v16h load_bT(const _Float16* rowT, int kb, int grp) {
  Frag f;
  f.q[0] = *(const uint4*)(rowT + kb + 8 * grp);
  f.q[1] = *(const uint4*)(rowT + kb + 16 + 8 * grp);
  return f.v;
}

__device__ __forceinline__ float gelu_exact(float x) {
  return 0.5f * x * (1.0f + erff(x * 0.70710678118654752440f));
}

// ---------------------------------------------------------------------------
// Prep: [nH][R][C] f32 -> [nH][C][R] f16, 32x32 LDS-tiled so both the global
// loads (128B/row) and the f16 stores (64B/row) are coalesced.
// Grid: (C/32, R/32, nH), block 256 (32x8).
// ---------------------------------------------------------------------------
__global__ __launch_bounds__(256) void transpose_convert_kernel(
    const float* __restrict__ src, _Float16* __restrict__ dst, int R, int C) {
  __shared__ float tile[32][33];  // +1 pad: no LDS bank conflicts
  const int hh = blockIdx.z;
  const float*    s = src + (size_t)hh * R * C;
  _Float16*       d = dst + (size_t)hh * R * C;
  const int c0 = blockIdx.x * 32;
  const int r0 = blockIdx.y * 32;
  const int tx = threadIdx.x & 31;
  const int ty = threadIdx.x >> 5;  // 0..7
#pragma unroll
  for (int i = 0; i < 32; i += 8)
    tile[ty + i][tx] = s[(size_t)(r0 + ty + i) * C + (c0 + tx)];
  __syncthreads();
#pragma unroll
  for (int i = 0; i < 32; i += 8)
    d[(size_t)(c0 + ty + i) * R + (r0 + tx)] = (_Float16)tile[tx][ty + i];
}

// ---------------------------------------------------------------------------
// Stage 1: per-head feature maps.
//   qf = |gelu(gelu(q W1) W2)|                               (kmode=0)
//   kf0 = |scalingD| * gelu(gelu(k W1) W2)
//   kf  = |kf0 + (kf0 @ interK) * scalingD2|                 (kmode=1)
// One block = (head h, 16 rows). 8 waves, each wave owns 16x16 WMMA tiles.
// ---------------------------------------------------------------------------
__global__ __launch_bounds__(256) void featmap_kernel(
    const float* __restrict__ in,          // [S][2048]
    const _Float16* __restrict__ W1T,      // [H][DHID][DHEAD]  (B^T of layer1)
    const _Float16* __restrict__ W2T,      // [H][DHEAD][DHID]  (B^T of layer2)
    const _Float16* __restrict__ interT,   // [H][DHEAD][DHEAD] (B^T) / null
    const float* __restrict__ scalingD,    // [H][DHEAD] / null
    const float* __restrict__ scalingD2,   // [H][DHEAD] / null
    _Float16* __restrict__ outAbs,         // [H][S][DHEAD]
    int kmode) {
  __shared__ __align__(16) _Float16 Ain[16 * DHEAD];
  __shared__ __align__(16) _Float16 T1[16 * DHID];
  __shared__ __align__(16) _Float16 T2[16 * DHEAD];

  const int h    = blockIdx.y;
  const int m0   = blockIdx.x * 16;
  const int tid  = threadIdx.x;
  const int wave = tid >> 5;
  const int lane = tid & 31;
  const int mrow = lane & 15;   // M (A/C row) and N (B/C col) index
  const int grp  = lane >> 4;

  // stage 16 input rows (head slice), f32 -> f16
  for (int i = tid; i < 16 * DHEAD; i += 256) {
    int m = i >> 7, d = i & 127;
    Ain[i] = (_Float16)in[(size_t)(m0 + m) * 2048 + h * DHEAD + d];
  }
  __syncthreads();

  // -------- layer1: [16,128] @ [128,256], gelu -> T1
  for (int half = 0; half < 2; ++half) {
    int nt = wave + half * 8;
    v8f acc = {};
    const _Float16* ar = Ain + mrow * DHEAD;
    const _Float16* br = W1T + ((size_t)h * DHID + nt * 16 + mrow) * DHEAD;
#pragma unroll
    for (int kb = 0; kb < DHEAD; kb += 32)
      acc = wmma32(load_a(ar, kb, grp), load_bT(br, kb, grp), acc);
#pragma unroll
    for (int r = 0; r < 8; ++r) {
      int m = r + 8 * grp;
      int n = nt * 16 + mrow;
      T1[m * DHID + n] = (_Float16)gelu_exact(acc[r]);
    }
  }
  __syncthreads();

  // -------- layer2: [16,256] @ [256,128], gelu (+scalingD for k)
  {
    int nt = wave;
    v8f acc = {};
    const _Float16* ar = T1 + mrow * DHID;
    const _Float16* br = W2T + ((size_t)h * DHEAD + nt * 16 + mrow) * DHID;
#pragma unroll
    for (int kb = 0; kb < DHID; kb += 32)
      acc = wmma32(load_a(ar, kb, grp), load_bT(br, kb, grp), acc);
#pragma unroll
    for (int r = 0; r < 8; ++r) {
      int m = r + 8 * grp;
      int n = nt * 16 + mrow;
      float g = gelu_exact(acc[r]);
      if (kmode) {
        T2[m * DHEAD + n] = (_Float16)(fabsf(scalingD[h * DHEAD + n]) * g);
      } else {
        outAbs[((size_t)h * S_LEN + m0 + m) * DHEAD + n] = (_Float16)fabsf(g);
      }
    }
  }
  if (!kmode) return;
  __syncthreads();

  // -------- layer3 (k only): kf = kf0 + (kf0 @ interK) * scalingD2
  {
    int nt = wave;
    v8f acc = {};
    const _Float16* ar = T2 + mrow * DHEAD;
    const _Float16* br = interT + ((size_t)h * DHEAD + nt * 16 + mrow) * DHEAD;
#pragma unroll
    for (int kb = 0; kb < DHEAD; kb += 32)
      acc = wmma32(load_a(ar, kb, grp), load_bT(br, kb, grp), acc);
#pragma unroll
    for (int r = 0; r < 8; ++r) {
      int m = r + 8 * grp;
      int n = nt * 16 + mrow;
      float kf = (float)T2[m * DHEAD + n] + acc[r] * scalingD2[h * DHEAD + n];
      outAbs[((size_t)h * S_LEN + m0 + m) * DHEAD + n] = (_Float16)fabsf(kf);
    }
  }
}

// ---------------------------------------------------------------------------
// Stage 2: fused attention. One block = (head h, 16 query rows).
// Pass 1: scores tile row via WMMA; read mask + sparse_w ONCE (the HBM-bound
//   335MB stream); store pre = m ? score+1e-6 : exp(sparse_w) (f16, 64KB LDS)
//   and accumulate masked row sums (per-lane regs + 8 ds_add_f32).
// Pass 2: nf = logaddexp(log(rowsum+1e-6), sparse_lse); rowscale = exp(-nf).
// Pass 3: out = diag(rowscale) @ (pre @ v): the row scale commutes with the
//   matmul, so A-fragments are raw b128 LDS reads of `pre` and the scale is
//   applied once per output element in the epilogue.
// ---------------------------------------------------------------------------
__global__ __launch_bounds__(256) void attn_kernel(
    const _Float16* __restrict__ qfA,       // [H][S][DHEAD]
    const _Float16* __restrict__ kfA,       // [H][S][DHEAD]
    const _Float16* __restrict__ vT,        // [H][DHEAD][S]
    const unsigned char* __restrict__ mask, // [H][S][S]
    const float* __restrict__ slse,         // [H][S]
    const float* __restrict__ sw,           // [H][S][S]
    float* __restrict__ out) {              // [S][2048]
  extern __shared__ __align__(16) _Float16 pre[];  // [16][S_LEN]
  __shared__ float rowsum[16];
  __shared__ float rowscale[16];

  const int h    = blockIdx.y;
  const int m0   = blockIdx.x * 16;
  const int tid  = threadIdx.x;
  const int wave = tid >> 5;
  const int lane = tid & 31;
  const int mrow = lane & 15;
  const int grp  = lane >> 4;

  if (tid < 16) rowsum[tid] = 0.0f;
  __syncthreads();

  const _Float16* qrow = qfA + ((size_t)h * S_LEN + m0 + mrow) * DHEAD;
  float rs[8] = {0.f, 0.f, 0.f, 0.f, 0.f, 0.f, 0.f, 0.f};

  // ---- pass 1: score tiles; mask+sparse fused, single HBM read of [S,S] data
  for (int jt = wave; jt < S_LEN / 16; jt += 8) {
    // prefetch next tile of the HBM-bound [S,S] streams into GL2
    if (jt + 8 < S_LEN / 16) {
      size_t pidx = ((size_t)h * S_LEN + (m0 + mrow)) * S_LEN + (jt + 8) * 16;
      __builtin_prefetch(sw + pidx, 0, 0);
      __builtin_prefetch(mask + pidx, 0, 0);
    }
    v8f acc = {};
    const _Float16* br = kfA + ((size_t)h * S_LEN + jt * 16 + mrow) * DHEAD;
#pragma unroll
    for (int kb = 0; kb < DHEAD; kb += 32)
      acc = wmma32(load_a(qrow, kb, grp), load_bT(br, kb, grp), acc);
#pragma unroll
    for (int r = 0; r < 8; ++r) {
      int m = r + 8 * grp;          // C layout: VGPR r, lanes 16-31 -> M+8
      int n = jt * 16 + mrow;       // C layout: N = lane&15
      size_t idx = ((size_t)h * S_LEN + (m0 + m)) * S_LEN + n;
      float sc = acc[r];
      bool mb  = mask[idx] != 0;
      float p  = mb ? (sc + 1e-6f) : __expf(sw[idx]);
      pre[m * S_LEN + n] = (_Float16)p;
      rs[r] += mb ? sc : 0.0f;
    }
  }
#pragma unroll
  for (int r = 0; r < 8; ++r) atomicAdd(&rowsum[r + 8 * grp], rs[r]);
  __syncthreads();

  // ---- pass 2: per-row normalization factor
  if (tid < 16) {
    float a  = logf(rowsum[tid] + 1e-6f);
    float b  = slse[(size_t)h * S_LEN + m0 + tid];
    float mx = fmaxf(a, b);
    float nf = mx + log1pf(expf(-fabsf(a - b)));  // logaddexp
    rowscale[tid] = expf(-nf);
  }
  __syncthreads();

  // ---- pass 3: out = attn @ v, one 16x16 d-tile per wave, K = S_LEN
  {
    int nt = wave;
    v8f acc = {};
    const _Float16* prow = pre + mrow * S_LEN;  // A row for this lane
    const _Float16* br = vT + ((size_t)h * DHEAD + nt * 16 + mrow) * S_LEN;
    for (int kb = 0; kb < S_LEN; kb += 32)
      acc = wmma32(load_a(prow, kb, grp), load_bT(br, kb, grp), acc);
#pragma unroll
    for (int r = 0; r < 8; ++r) {
      int m = r + 8 * grp;
      int d = nt * 16 + mrow;
      out[(size_t)(m0 + m) * 2048 + h * DHEAD + d] = acc[r] * rowscale[m];
    }
  }
}

// ---------------------------------------------------------------------------
// Workspace layout (bytes):
//   qfA   [H][S][128] f16   @ 0          (8,388,608)
//   kfA   [H][S][128] f16   @ 8388608    (8,388,608)
//   vT    [H][128][S] f16   @ 16777216   (8,388,608)
//   W1qT  [H][256][128] f16 @ 25165824   (1,048,576)
//   W1kT                    @ 26214400
//   W2qT  [H][128][256] f16 @ 27262976
//   W2kT                    @ 28311552
//   ikT   [H][128][128] f16 @ 29360128   (524,288)   total ~29.9 MB
// ---------------------------------------------------------------------------
static void launch_transpose(const float* s, _Float16* d, int nH, int R, int C,
                             hipStream_t stream) {
  dim3 grid(C / 32, R / 32, nH);
  transpose_convert_kernel<<<grid, 256, 0, stream>>>(s, d, R, C);
}

extern "C" void kernel_launch(void* const* d_in, const int* in_sizes, int n_in,
                              void* d_out, int out_size, void* d_ws, size_t ws_size,
                              hipStream_t stream) {
  (void)in_sizes; (void)n_in; (void)out_size; (void)ws_size;
  const float* q    = (const float*)d_in[1];
  const float* k    = (const float*)d_in[2];
  const float* v    = (const float*)d_in[3];
  const unsigned char* mask = (const unsigned char*)d_in[4];   // bool mask
  const float* slse = (const float*)d_in[5];
  const float* sw   = (const float*)d_in[6];
  const float* kq1  = (const float*)d_in[8];
  const float* kk1  = (const float*)d_in[9];
  const float* kq2  = (const float*)d_in[10];
  const float* kk2  = (const float*)d_in[11];
  const float* ik   = (const float*)d_in[12];
  const float* sD   = (const float*)d_in[13];
  const float* sD2  = (const float*)d_in[14];

  char* ws = (char*)d_ws;
  _Float16* qfA  = (_Float16*)(ws + 0);
  _Float16* kfA  = (_Float16*)(ws + 8388608);
  _Float16* vT   = (_Float16*)(ws + 16777216);
  _Float16* W1qT = (_Float16*)(ws + 25165824);
  _Float16* W1kT = (_Float16*)(ws + 26214400);
  _Float16* W2qT = (_Float16*)(ws + 27262976);
  _Float16* W2kT = (_Float16*)(ws + 28311552);
  _Float16* ikT  = (_Float16*)(ws + 29360128);

  // prep: transpose+convert weights and v into WMMA B^T f16 layouts
  launch_transpose(kq1, W1qT, NHEAD, DHEAD, DHID, stream);
  launch_transpose(kk1, W1kT, NHEAD, DHEAD, DHID, stream);
  launch_transpose(kq2, W2qT, NHEAD, DHID, DHEAD, stream);
  launch_transpose(kk2, W2kT, NHEAD, DHID, DHEAD, stream);
  launch_transpose(ik,  ikT,  NHEAD, DHEAD, DHEAD, stream);
  // v [S][2048] -> [2048][S]; row (h*128+d) == vT[h][d][s]
  launch_transpose(v, vT, 1, S_LEN, 2048, stream);

  dim3 grid(S_LEN / 16, NHEAD);
  featmap_kernel<<<grid, 256, 0, stream>>>(q, W1qT, W2qT, nullptr, nullptr,
                                           nullptr, qfA, /*kmode=*/0);
  featmap_kernel<<<grid, 256, 0, stream>>>(k, W1kT, W2kT, ikT, sD, sD2, kfA,
                                           /*kmode=*/1);
  // 16 x 2048 f16 `pre` buffer in dynamic LDS (64 KB of the 320 KB WGP LDS)
  attn_kernel<<<grid, 256, 16 * S_LEN * sizeof(_Float16), stream>>>(
      qfA, kfA, vT, mask, slse, sw, (float*)d_out);
}
